// AttentionBlock_84808424227130
// MI455X (gfx1250) — matrix operand order repaired and checked by reading, not compiled, not run on hardware
//
#include <hip/hip_runtime.h>
#include <cstdint>
#include <cstddef>

// ---------------------------------------------------------------------------
// Types for CDNA5 WMMA (wave32) + TDM
// ---------------------------------------------------------------------------
typedef _Float16 f16;
typedef __attribute__((ext_vector_type(16))) _Float16 v16h;
typedef __attribute__((ext_vector_type(8)))  _Float16 v8h;
typedef __attribute__((ext_vector_type(8)))  float    v8f;
typedef __attribute__((ext_vector_type(4)))  unsigned int u32x4;
typedef __attribute__((ext_vector_type(8)))  int          i32x8;
typedef __attribute__((ext_vector_type(4)))  int          i32x4;

// Problem constants
#define BB      8
#define CC      512
#define TT      1024          // H*W
#define HEADS   8
#define CH      64
#define SS      77
#define SPAD    1120          // (77+1024) padded to 35*32
#define GROUPS  32
#define NCHUNK  35            // SPAD/32

// Workspace layout (bytes, all 256-aligned)
#define OFF_QKVW  ((size_t)0)                 // 1536*512 f16
#define OFF_EKVW  ((size_t)1572864)           // 1024*512 f16
#define OFF_PROJW ((size_t)2621440)           // 512*512  f16
#define OFF_NRM   ((size_t)3145728)           // [B][T][C] f16
#define OFF_ENC   ((size_t)11534336)          // [B][80][C] f16 (padded S)
#define OFF_Q     ((size_t)12189696)          // [64][1024][64] f16 (pre-scaled)
#define OFF_K     ((size_t)20578304)          // [64][1120][64] f16 (pre-scaled)
#define OFF_V     ((size_t)29753344)          // [64][64][1120] f16
#define OFF_O     ((size_t)38928384)          // [B][T][C] f16 attention output

#define QK_SCALE 0.35355339059327373f         // 1/sqrt(sqrt(64))

// ---------------------------------------------------------------------------
// WMMA helpers
// ---------------------------------------------------------------------------
__device__ __forceinline__ v8f wmma_f16(v16h a, v16h b, v8f c) {
  return __builtin_amdgcn_wmma_f32_16x16x32_f16(
      /*neg_a=*/false, a, /*neg_b=*/false, b,
      /*c_mod=*/(short)0, c, /*reuse_a=*/false, /*reuse_b=*/false);
}

__device__ __forceinline__ v16h pack16(v8h lo, v8h hi) {
  v16h r;
#pragma unroll
  for (int i = 0; i < 8; ++i) { r[i] = lo[i]; r[8 + i] = hi[i]; }
  return r;
}

// A-matrix fragment (16xK slice, row-major [m][k], 16-bit):
// lane<16: row=lane,    K = kb+0..7  and kb+16..23
// lane>=16: row=lane-16, K = kb+8..15 and kb+24..31
__device__ __forceinline__ v16h load_a_frag(const f16* __restrict__ base, int ld,
                                            int m0, int kb, int lane) {
  int row  = m0 + (lane & 15);
  int koff = (lane >> 4) * 8;
  const f16* p = base + (size_t)row * ld + kb + koff;
  v8h lo = *(const v8h*)(p);
  v8h hi = *(const v8h*)(p + 16);
  return pack16(lo, hi);
}

// B-matrix fragment (Kx16 slice, stored as [n][k] contiguous-k):
// lane<16: col=lane, K = kb+0..7 and kb+16..23 ; lane>=16: col=lane-16, K = kb+8..15 and kb+24..31
__device__ __forceinline__ v16h load_b_frag(const f16* __restrict__ base, int ld,
                                            int n0, int kb, int lane) {
  int col  = n0 + (lane & 15);
  int koff = (lane >> 4) * 8;
  const f16* p = base + (size_t)col * ld + kb + koff;
  v8h lo = *(const v8h*)(p);
  v8h hi = *(const v8h*)(p + 16);
  return pack16(lo, hi);
}

// ---------------------------------------------------------------------------
// Tensor Data Mover: 2D tile global -> LDS (f16 elements, data_size=2B)
// Descriptor per CDNA5 ISA D# groups 0/1; groups 2/3 and
// the extra descriptor slot are zero (<=2D tensor). dims/tiles/strides in
// elements. 6-arg builtin form (clang-23 / amdgpu-toolchain).
// ---------------------------------------------------------------------------
__device__ __forceinline__ void tdm_load_2d_f16(uint32_t lds_addr, const void* gaddr,
                                                uint32_t dim0, uint32_t dim1,
                                                uint32_t tile0, uint32_t tile1,
                                                uint32_t stride0) {
  uint64_t ga = (uint64_t)(uintptr_t)gaddr;
  u32x4 g0;
  g0[0] = 1u;                                            // count=1, user load
  g0[1] = lds_addr;                                      // LDS byte address
  g0[2] = (uint32_t)ga;                                  // global_addr[31:0]
  g0[3] = (uint32_t)((ga >> 32) & 0x01FFFFFFu) | (2u << 30); // ga[56:32], type=2
  i32x8 g1;
  g1[0] = (int)(1u << 16);                               // data_size=1 (2 bytes)
  g1[1] = (int)(dim0 << 16);                             // tensor_dim0[15:0]
  g1[2] = (int)((dim0 >> 16) | (dim1 << 16));            // dim0 hi | dim1 lo
  g1[3] = (int)((dim1 >> 16) | (tile0 << 16));           // dim1 hi | tile_dim0
  g1[4] = (int)(tile1 & 0xFFFFu);                        // tile_dim1 (tile_dim2=0)
  g1[5] = (int)stride0;                                  // tensor_dim0_stride lo
  g1[6] = 0;                                             // stride0 hi | stride1 lo
  g1[7] = 0;
  i32x4 z4 = {0, 0, 0, 0};
  i32x8 z8 = {0, 0, 0, 0, 0, 0, 0, 0};
  __builtin_amdgcn_tensor_load_to_lds(g0, g1, z4, z4, z8, 0);
}

// ---------------------------------------------------------------------------
// Utility kernels
// ---------------------------------------------------------------------------
__global__ void cvt_f32_to_f16_kernel(const float* __restrict__ in,
                                      f16* __restrict__ out, int n) {
  int i = blockIdx.x * blockDim.x + threadIdx.x;
  if (i < n) out[i] = (f16)in[i];
}

__global__ void zero_b16_kernel(uint32_t* __restrict__ p, int n_u32) {
  int i = blockIdx.x * blockDim.x + threadIdx.x;
  if (i < n_u32) p[i] = 0u;
}

// encoder_out [B][C][S] f32 -> enc_tc [B][80][C] f16 (zero padded s>=77)
__global__ void enc_transpose_kernel(const float* __restrict__ enc,
                                     f16* __restrict__ out) {
  int i = blockIdx.x * blockDim.x + threadIdx.x;      // over B*80*C
  if (i >= BB * 80 * CC) return;
  int c = i % CC;
  int s = (i / CC) % 80;
  int b = i / (80 * CC);
  float v = (s < SS) ? enc[((size_t)b * CC + c) * SS + s] : 0.0f;
  out[i] = (f16)v;
}

// ---------------------------------------------------------------------------
// GroupNorm32 (affine) -> transposed f16 activations [b][t][c]
// ---------------------------------------------------------------------------
__global__ void groupnorm_kernel(const float* __restrict__ x,
                                 const float* __restrict__ gamma,
                                 const float* __restrict__ beta,
                                 f16* __restrict__ nrm_tc) {
  const int b = blockIdx.x / GROUPS;
  const int g = blockIdx.x % GROUPS;
  const int CG = CC / GROUPS;          // 16 channels per group
  const float* xp = x + ((size_t)b * CC + g * CG) * TT;

  __shared__ float s_sum[256];
  __shared__ float s_sq[256];
  float sum = 0.f, sq = 0.f;
  for (int i = threadIdx.x; i < CG * TT; i += 256) {
    float v = xp[i];
    sum += v; sq += v * v;
  }
  s_sum[threadIdx.x] = sum;
  s_sq[threadIdx.x]  = sq;
  __syncthreads();
  for (int off = 128; off > 0; off >>= 1) {
    if ((int)threadIdx.x < off) {
      s_sum[threadIdx.x] += s_sum[threadIdx.x + off];
      s_sq[threadIdx.x]  += s_sq[threadIdx.x + off];
    }
    __syncthreads();
  }
  const float inv_n = 1.0f / (float)(CG * TT);
  float mu  = s_sum[0] * inv_n;
  float var = s_sq[0] * inv_n - mu * mu;
  float inv = rsqrtf(var + 1e-5f);

  for (int i = threadIdx.x; i < CG * TT; i += 256) {
    int ch = i >> 10;
    int t  = i & (TT - 1);
    int cg = g * CG + ch;
    float v = (xp[i] - mu) * inv * gamma[cg] + beta[cg];
    nrm_tc[((size_t)b * TT + t) * CC + cg] = (f16)v;
  }
}

// ---------------------------------------------------------------------------
// qkv GEMM: [1536x512] x [512x1024] per batch; wave -> 16x64 tile, software
// pipelined (register double buffering). Scatters into Q/K/V layouts.
// grid = B * 96 * 16 blocks, 32 threads
// ---------------------------------------------------------------------------
__global__ void __launch_bounds__(32)
qkv_gemm_kernel(const f16* __restrict__ W, const f16* __restrict__ Bm,
                const float* __restrict__ bias,
                f16* __restrict__ Qb, f16* __restrict__ Kb, f16* __restrict__ Vb) {
  const int lane = threadIdx.x;
  const int id = blockIdx.x;
  const int b  = id / 1536;
  const int r  = id % 1536;
  const int m0 = (r >> 4) * 16;
  const int n0 = (r & 15) * 64;
  const f16* Bp = Bm + (size_t)b * TT * CC;

  v8f acc[4] = {};
  v16h a_cur = load_a_frag(W, CC, m0, 0, lane);
  v16h b_cur[4];
#pragma unroll
  for (int s = 0; s < 4; ++s) b_cur[s] = load_b_frag(Bp, CC, n0 + s * 16, 0, lane);

  for (int kb = 0; kb < CC; kb += 32) {
    v16h a_nxt = a_cur;
    v16h b_nxt[4] = {b_cur[0], b_cur[1], b_cur[2], b_cur[3]};
    const int kn = kb + 32;
    if (kn < CC) {                         // prefetch next chunk
      a_nxt = load_a_frag(W, CC, m0, kn, lane);
#pragma unroll
      for (int s = 0; s < 4; ++s) b_nxt[s] = load_b_frag(Bp, CC, n0 + s * 16, kn, lane);
    }
#pragma unroll
    for (int s = 0; s < 4; ++s) acc[s] = wmma_f16(a_cur, b_cur[s], acc[s]);
    a_cur = a_nxt;
#pragma unroll
    for (int s = 0; s < 4; ++s) b_cur[s] = b_nxt[s];
  }

  const int head  = m0 / (3 * CH);
  const int inner = m0 % (3 * CH);
  const int which = inner / CH;            // 0=q 1=k 2=v (uniform over tile)
  const int c0    = inner % CH;
  const int bh    = b * HEADS + head;
  const int half  = lane >> 4, col_l = lane & 15;

#pragma unroll
  for (int s = 0; s < 4; ++s) {
    int t = n0 + s * 16 + col_l;
#pragma unroll
    for (int v = 0; v < 8; ++v) {
      int m = v + 8 * half;
      float val = acc[s][v] + bias[m0 + m];
      int c = c0 + m;
      if (which == 0)
        Qb[((size_t)bh * TT + t) * CH + c] = (f16)(val * QK_SCALE);
      else if (which == 1)
        Kb[((size_t)bh * SPAD + (SS + t)) * CH + c] = (f16)(val * QK_SCALE);
      else
        Vb[((size_t)bh * CH + c) * SPAD + (SS + t)] = (f16)val;
    }
  }
}

// ---------------------------------------------------------------------------
// ekv GEMM: [1024x512] x [512x77] per batch; wave -> 16x16 tile (N padded 80).
// grid = B * 64 * 5 blocks, 32 threads
// ---------------------------------------------------------------------------
__global__ void __launch_bounds__(32)
ekv_gemm_kernel(const f16* __restrict__ W, const f16* __restrict__ E,
                const float* __restrict__ bias,
                f16* __restrict__ Kb, f16* __restrict__ Vb) {
  const int lane = threadIdx.x;
  const int id = blockIdx.x;
  const int b  = id / 320;
  const int r  = id % 320;
  const int m0 = (r / 5) * 16;
  const int n0 = (r % 5) * 16;
  const f16* Ep = E + (size_t)b * 80 * CC;

  v8f acc = {};
  v16h a_cur = load_a_frag(W, CC, m0, 0, lane);
  v16h b_cur = load_b_frag(Ep, CC, n0, 0, lane);
  for (int kb = 0; kb < CC; kb += 32) {
    v16h a_nxt = a_cur, b_nxt = b_cur;
    const int kn = kb + 32;
    if (kn < CC) {
      a_nxt = load_a_frag(W, CC, m0, kn, lane);
      b_nxt = load_b_frag(Ep, CC, n0, kn, lane);
    }
    acc = wmma_f16(a_cur, b_cur, acc);
    a_cur = a_nxt; b_cur = b_nxt;
  }

  const int head  = m0 / (2 * CH);
  const int inner = m0 % (2 * CH);
  const int which = inner / CH;            // 0=ek 1=ev
  const int c0    = inner % CH;
  const int bh    = b * HEADS + head;
  const int half  = lane >> 4, col_l = lane & 15;
  const int s = n0 + col_l;

  if (s < SS) {
#pragma unroll
    for (int v = 0; v < 8; ++v) {
      int m = v + 8 * half;
      float val = acc[v] + bias[m0 + m];
      int c = c0 + m;
      if (which == 0)
        Kb[((size_t)bh * SPAD + s) * CH + c] = (f16)(val * QK_SCALE);
      else
        Vb[((size_t)bh * CH + c) * SPAD + s] = (f16)val;
    }
  }
}

// ---------------------------------------------------------------------------
// Flash attention: one wave per 16 query rows of one (b,head).
// K/V tiles staged in LDS by the Tensor Data Mover (double buffered,
// synchronized with s_wait_tensorcnt); QK^T and P*V^T via WMMA; online
// softmax with wave32 shfl_xor butterflies.
// grid = 64 * 64 blocks, 32 threads
// ---------------------------------------------------------------------------
__global__ void __launch_bounds__(32)
attn_kernel(const f16* __restrict__ Qb, const f16* __restrict__ Kb,
            const f16* __restrict__ Vb, const int* __restrict__ capt,
            f16* __restrict__ Ob) {
  const int lane = threadIdx.x;
  const int bh = blockIdx.x >> 6;
  const int t0 = (blockIdx.x & 63) * 16;
  const int b = bh >> 3, head = bh & 7;
  const f16* Qp = Qb + (size_t)bh * TT * CH;
  const f16* Kp = Kb + (size_t)bh * SPAD * CH;
  const f16* Vp = Vb + (size_t)bh * CH * SPAD;
  const int* cr = capt + (size_t)(bh & (BB - 1)) * SS;   // tile() head-major bug

  __shared__ __align__(16) f16 lds_k[2][32 * CH];   // [s_local][c]
  __shared__ __align__(16) f16 lds_v[2][CH * 32];   // [c][s_local]
  __shared__ __align__(16) f16 lds_p[16 * 40];      // P tile, stride 40 halves

  const int half = lane >> 4, col_l = lane & 15;

  // Q A-fragments for all of K=64, kept in registers for the whole loop
  v16h qa0 = load_a_frag(Qp, CH, t0, 0, lane);
  v16h qa1 = load_a_frag(Qp, CH, t0, 32, lane);

  float m_run[8], l_run[8];
#pragma unroll
  for (int v = 0; v < 8; ++v) { m_run[v] = -1e30f; l_run[v] = 0.f; }
  v8f O[4] = {};

  // TDM prologue: chunk 0 into buffer 0
  tdm_load_2d_f16((uint32_t)(uintptr_t)&lds_k[0][0], Kp, CH, 32, CH, 32, CH);
  tdm_load_2d_f16((uint32_t)(uintptr_t)&lds_v[0][0], Vp, 32, CH, 32, CH, SPAD);

  for (int ci = 0; ci < NCHUNK; ++ci) {
    const int s0 = ci * 32;
    const int cur = ci & 1;
    if (ci + 1 < NCHUNK) {                 // TDM prefetch of next chunk
      const int s1 = s0 + 32;
      tdm_load_2d_f16((uint32_t)(uintptr_t)&lds_k[cur ^ 1][0],
                      Kp + (size_t)s1 * CH, CH, 32, CH, 32, CH);
      tdm_load_2d_f16((uint32_t)(uintptr_t)&lds_v[cur ^ 1][0],
                      Vp + s1, 32, CH, 32, CH, SPAD);
      __builtin_amdgcn_s_wait_tensorcnt(2);   // older pair (current chunk) done
    } else {
      __builtin_amdgcn_s_wait_tensorcnt(0);
    }
    asm volatile("" ::: "memory");            // no LDS-read CSE across TDM rewrite
    const f16* kk = lds_k[cur];
    const f16* vv = lds_v[cur];

#pragma unroll
    for (int sub = 0; sub < 2; ++sub) {
      v8f acc = {};
      acc = wmma_f16(qa0, load_b_frag(kk, CH, sub * 16, 0, lane), acc);
      acc = wmma_f16(qa1, load_b_frag(kk, CH, sub * 16, 32, lane), acc);

      const int col = s0 + sub * 16 + col_l;
      float addm;
      if (col < SS)             addm = cr[col] ? 0.0f : -10000.0f;
      else if (col < SS + TT)   addm = 0.0f;
      else                      addm = -1e30f;          // padding

      float tmax[8];
#pragma unroll
      for (int v = 0; v < 8; ++v) { acc[v] += addm; tmax[v] = acc[v]; }
#pragma unroll
      for (int off = 1; off < 16; off <<= 1)
#pragma unroll
        for (int v = 0; v < 8; ++v)
          tmax[v] = fmaxf(tmax[v], __shfl_xor(tmax[v], off, 32));

      float sc[8], rs[8];
#pragma unroll
      for (int v = 0; v < 8; ++v) {
        float mnew = fmaxf(m_run[v], tmax[v]);
        sc[v] = __expf(m_run[v] - mnew);
        float p = __expf(acc[v] - mnew);
        m_run[v] = mnew;
        acc[v] = p;
        rs[v] = p;
      }
#pragma unroll
      for (int off = 1; off < 16; off <<= 1)
#pragma unroll
        for (int v = 0; v < 8; ++v)
          rs[v] += __shfl_xor(rs[v], off, 32);
#pragma unroll
      for (int v = 0; v < 8; ++v) l_run[v] = l_run[v] * sc[v] + rs[v];
#pragma unroll
      for (int n = 0; n < 4; ++n)
#pragma unroll
        for (int v = 0; v < 8; ++v) O[n][v] *= sc[v];

      // P tile -> LDS in [m][k] layout for the A-fragment of P*V^T
#pragma unroll
      for (int v = 0; v < 8; ++v)
        lds_p[(v + 8 * half) * 40 + sub * 16 + col_l] = (f16)acc[v];
    }

    // P A-fragment (16x32) from LDS, then P*V^T WMMAs over K=32
    v16h pa;
    {
      int row  = col_l;
      int koff = half * 8;
      v8h lo = *(const v8h*)(lds_p + row * 40 + koff);
      v8h hi = *(const v8h*)(lds_p + row * 40 + koff + 16);
      pa = pack16(lo, hi);
    }
#pragma unroll
    for (int n = 0; n < 4; ++n) {
      v16h vb = load_b_frag(vv, 32, n * 16, 0, lane);
      O[n] = wmma_f16(pa, vb, O[n]);
    }
  }

  // normalize and write [b][t][head*64+c] f16
#pragma unroll
  for (int n = 0; n < 4; ++n) {
    int c = head * CH + n * 16 + col_l;
#pragma unroll
    for (int v = 0; v < 8; ++v) {
      int t = t0 + v + 8 * half;
      float val = O[n][v] / l_run[v];
      Ob[((size_t)b * TT + t) * CC + c] = (f16)val;
    }
  }
}

// ---------------------------------------------------------------------------
// proj GEMM + bias + residual: out = proj_w @ attn_out + proj_b + x
// wave -> 16x64 fp32 output tile, software pipelined.
// grid = B * 32 * 16 blocks, 32 threads
// ---------------------------------------------------------------------------
__global__ void __launch_bounds__(32)
proj_gemm_kernel(const f16* __restrict__ W, const f16* __restrict__ Bm,
                 const float* __restrict__ bias, const float* __restrict__ x,
                 float* __restrict__ out) {
  const int lane = threadIdx.x;
  const int id = blockIdx.x;
  const int b  = id / 512;
  const int r  = id % 512;
  const int m0 = (r >> 4) * 16;
  const int n0 = (r & 15) * 64;
  const f16* Bp = Bm + (size_t)b * TT * CC;

  v8f acc[4] = {};
  v16h a_cur = load_a_frag(W, CC, m0, 0, lane);
  v16h b_cur[4];
#pragma unroll
  for (int s = 0; s < 4; ++s) b_cur[s] = load_b_frag(Bp, CC, n0 + s * 16, 0, lane);

  for (int kb = 0; kb < CC; kb += 32) {
    v16h a_nxt = a_cur;
    v16h b_nxt[4] = {b_cur[0], b_cur[1], b_cur[2], b_cur[3]};
    const int kn = kb + 32;
    if (kn < CC) {
      a_nxt = load_a_frag(W, CC, m0, kn, lane);
#pragma unroll
      for (int s = 0; s < 4; ++s) b_nxt[s] = load_b_frag(Bp, CC, n0 + s * 16, kn, lane);
    }
#pragma unroll
    for (int s = 0; s < 4; ++s) acc[s] = wmma_f16(a_cur, b_cur[s], acc[s]);
    a_cur = a_nxt;
#pragma unroll
    for (int s = 0; s < 4; ++s) b_cur[s] = b_nxt[s];
  }

  const int half = lane >> 4, col_l = lane & 15;
#pragma unroll
  for (int s = 0; s < 4; ++s) {
    int t = n0 + s * 16 + col_l;
#pragma unroll
    for (int v = 0; v < 8; ++v) {
      int o = m0 + v + 8 * half;
      size_t idx = ((size_t)b * CC + o) * TT + t;
      out[idx] = acc[s][v] + bias[o] + x[idx];
    }
  }
}

// ---------------------------------------------------------------------------
// Launch
// ---------------------------------------------------------------------------
extern "C" void kernel_launch(void* const* d_in, const int* in_sizes, int n_in,
                              void* d_out, int out_size, void* d_ws, size_t ws_size,
                              hipStream_t stream) {
  (void)in_sizes; (void)n_in; (void)out_size; (void)ws_size;
  const float* x      = (const float*)d_in[0];
  const float* enc    = (const float*)d_in[1];
  const int*   capt   = (const int*)d_in[2];
  const float* nsc    = (const float*)d_in[3];
  const float* nbi    = (const float*)d_in[4];
  const float* qkv_w  = (const float*)d_in[5];
  const float* qkv_b  = (const float*)d_in[6];
  const float* ekv_w  = (const float*)d_in[7];
  const float* ekv_b  = (const float*)d_in[8];
  const float* proj_w = (const float*)d_in[9];
  const float* proj_b = (const float*)d_in[10];
  float* out = (float*)d_out;

  char* ws = (char*)d_ws;
  f16* qkv_w_h  = (f16*)(ws + OFF_QKVW);
  f16* ekv_w_h  = (f16*)(ws + OFF_EKVW);
  f16* proj_w_h = (f16*)(ws + OFF_PROJW);
  f16* nrm_tc   = (f16*)(ws + OFF_NRM);
  f16* enc_tc   = (f16*)(ws + OFF_ENC);
  f16* Qb       = (f16*)(ws + OFF_Q);
  f16* Kb       = (f16*)(ws + OFF_K);
  f16* Vb       = (f16*)(ws + OFF_V);
  f16* Ob       = (f16*)(ws + OFF_O);

  // weight conversions
  cvt_f32_to_f16_kernel<<<(1536 * 512 + 255) / 256, 256, 0, stream>>>(qkv_w, qkv_w_h, 1536 * 512);
  cvt_f32_to_f16_kernel<<<(1024 * 512 + 255) / 256, 256, 0, stream>>>(ekv_w, ekv_w_h, 1024 * 512);
  cvt_f32_to_f16_kernel<<<(512 * 512 + 255) / 256, 256, 0, stream>>>(proj_w, proj_w_h, 512 * 512);

  // encoder transpose (padded)
  enc_transpose_kernel<<<(BB * 80 * CC + 255) / 256, 256, 0, stream>>>(enc, enc_tc);

  // zero K and V buffers (contiguous) so padding rows/cols are 0
  {
    int n_u32 = (int)((2u * 64u * SPAD * CH * 2u) / 4u);   // bytes/4
    zero_b16_kernel<<<(n_u32 + 255) / 256, 256, 0, stream>>>((uint32_t*)Kb, n_u32);
  }

  // GroupNorm32 -> transposed f16
  groupnorm_kernel<<<BB * GROUPS, 256, 0, stream>>>(x, nsc, nbi, nrm_tc);

  // qkv projection (scatters into Q/K/V attention layouts)
  qkv_gemm_kernel<<<BB * 96 * 16, 32, 0, stream>>>(qkv_w_h, nrm_tc, qkv_b, Qb, Kb, Vb);

  // encoder k/v projection
  ekv_gemm_kernel<<<BB * 64 * 5, 32, 0, stream>>>(ekv_w_h, enc_tc, ekv_b, Kb, Vb);

  // attention (TDM-staged K/V)
  attn_kernel<<<64 * 64, 32, 0, stream>>>(Qb, Kb, Vb, capt, Ob);

  // output projection + residual
  proj_gemm_kernel<<<BB * 32 * 16, 32, 0, stream>>>(proj_w_h, Ob, proj_b, x, out);
}
